// QuantFastdLLMAttention_71382356459967
// MI455X (gfx1250) — 8-bit matrix operand order repaired and checked by reading, compiled, not run on hardware
//
#include <hip/hip_runtime.h>
#include <stdint.h>


typedef int v8i __attribute__((ext_vector_type(8)));

#define DEVFN static __device__ __forceinline__

DEVFN v8i wmma_iu8(v8i a, v8i b, v8i c) {
  return __builtin_amdgcn_wmma_i32_16x16x64_iu8(true, a, true, b, c, false, false);
}

DEVFN v8i load_a_tile(const void* row64, int hi) {
  const char* p = (const char*)row64;
  v8i r;
#pragma unroll
  for (int v = 0; v < 8; ++v)
    r[v] = *(const int*)(p + ((v & 1) * 4 + (v >> 1) * 16 + hi * 8));
  return r;
}

DEVFN v8i load_b_tile(const void* rowT64, int hi) {
  const char* p = (const char*)rowT64;
  v8i r;
#pragma unroll
  for (int v = 0; v < 8; ++v)
    r[v] = *(const int*)(p + ((v & 1) * 4 + (v >> 1) * 16 + hi * 8));
  return r;
}

DEVFN int clampi(int v, int lo, int hi) { return v < lo ? lo : (v > hi ? hi : v); }

__global__ void zero_amax_kernel(float* amax) {
  if (threadIdx.x < 16) amax[threadIdx.x] = 0.0f;
}

__global__ void absmax_kernel(const float* __restrict__ x, size_t n,
                              float* __restrict__ slot) {
  float m = 0.0f;
  for (size_t i = (size_t)blockIdx.x * blockDim.x + threadIdx.x; i < n;
       i += (size_t)gridDim.x * blockDim.x)
    m = fmaxf(m, fabsf(x[i]));
#pragma unroll
  for (int off = 16; off; off >>= 1) m = fmaxf(m, __shfl_xor(m, off, 32));
  __shared__ float red[8];
  if ((threadIdx.x & 31) == 0) red[threadIdx.x >> 5] = m;
  __syncthreads();
  if (threadIdx.x == 0) {
    float bm = red[0];
    for (int w = 1; w < (int)(blockDim.x >> 5); ++w) bm = fmaxf(bm, red[w]);
    atomicMax((unsigned int*)slot, __float_as_uint(bm));
  }
}

__global__ void quant_kernel(const float* __restrict__ x, size_t n,
                             const float* __restrict__ slot,
                             int8_t* __restrict__ q) {
  float inv = 1.0f / fmaxf(slot[0] * (1.0f / 127.0f), 1e-8f);
  for (size_t i = (size_t)blockIdx.x * blockDim.x + threadIdx.x; i < n;
       i += (size_t)gridDim.x * blockDim.x)
    q[i] = (int8_t)clampi(__float2int_rn(x[i] * inv), -128, 127);
}

__global__ void gemm_i8_kernel(const int8_t* __restrict__ A,
                               const int8_t* __restrict__ Bm,
                               float* __restrict__ C, int M, int N, int K,
                               const float* __restrict__ amaxA,
                               const float* __restrict__ amaxB) {
  __shared__ char As[128 * 64];
  __shared__ char Bt[64 * 64];
  const int wave = threadIdx.x >> 5, lane = threadIdx.x & 31;
  const int hi = lane >> 4, ln = lane & 15;
  const int bm = blockIdx.y * 128, bn = blockIdx.x * 64;
  v8i acc[4] = {};
  for (int kc = 0; kc < K; kc += 64) {
    {
      int row = threadIdx.x >> 1, half = threadIdx.x & 1;
      const char* src = (const char*)A + (size_t)(bm + row) * K + kc + half * 32;
      char* dst = As + row * 64 + half * 32;
      *(uint4*)dst = *(const uint4*)src;
      *(uint4*)(dst + 16) = *(const uint4*)(src + 16);
      __builtin_prefetch(src + 64, 0, 3);
    }
    {
      int k = threadIdx.x & 63, ng = threadIdx.x >> 6;
      const char* src = (const char*)Bm + (size_t)(kc + k) * N + bn + ng * 16;
      uint4 w = *(const uint4*)src;
      const char* wb = (const char*)&w;
#pragma unroll
      for (int j = 0; j < 16; ++j) Bt[(ng * 16 + j) * 64 + k] = wb[j];
      __builtin_prefetch(src + (size_t)64 * N, 0, 3);
    }
    __syncthreads();
    v8i a = load_a_tile(As + (wave * 16 + ln) * 64, hi);
#pragma unroll
    for (int t = 0; t < 4; ++t)
      acc[t] = wmma_iu8(a, load_b_tile(Bt + (t * 16 + ln) * 64, hi), acc[t]);
    __syncthreads();
  }
  float sc = fmaxf(amaxA[0] * (1.0f / 127.0f), 1e-8f) *
             fmaxf(amaxB[0] * (1.0f / 127.0f), 1e-8f);
  int row0 = bm + wave * 16 + hi * 8;
#pragma unroll
  for (int t = 0; t < 4; ++t)
#pragma unroll
    for (int r = 0; r < 8; ++r)
      C[(size_t)(row0 + r) * N + bn + t * 16 + ln] = (float)acc[t][r] * sc;
}

__global__ void rope_absmax_kernel(float* __restrict__ buf,
                                   const float* __restrict__ cosb,
                                   const float* __restrict__ sinb, int nh,
                                   float* __restrict__ slot) {
  const int SEQ = 2048;
  const int Cc = nh * 128;
  size_t total = (size_t)4096 * nh * 64;
  float mx = 0.0f;
  for (size_t i = (size_t)blockIdx.x * blockDim.x + threadIdx.x; i < total;
       i += (size_t)gridDim.x * blockDim.x) {
    int d = (int)(i & 63);
    int hh = (int)((i >> 6) % nh);
    size_t row = i / ((size_t)nh << 6);
    int s = (int)(row & (SEQ - 1));
    float* p = buf + row * Cc + hh * 128;
    float x1 = p[d], x2 = p[d + 64];
    float c = cosb[s * 128 + d], sn = sinb[s * 128 + d];
    float o1 = x1 * c - x2 * sn;
    float o2 = x2 * c + x1 * sn;
    p[d] = o1;
    p[d + 64] = o2;
    mx = fmaxf(mx, fmaxf(fabsf(o1), fabsf(o2)));
  }
#pragma unroll
  for (int off = 16; off; off >>= 1) mx = fmaxf(mx, __shfl_xor(mx, off, 32));
  __shared__ float red[8];
  if ((threadIdx.x & 31) == 0) red[threadIdx.x >> 5] = mx;
  __syncthreads();
  if (threadIdx.x == 0) {
    float bm = red[0];
    for (int w = 1; w < (int)(blockDim.x >> 5); ++w) bm = fmaxf(bm, red[w]);
    atomicMax((unsigned int*)slot, __float_as_uint(bm));
  }
}

__global__ void quantT_head_kernel(const float* __restrict__ x,
                                   const float* __restrict__ slot, int nh,
                                   int8_t* __restrict__ out) {
  size_t total = (size_t)2 * nh * 2048 * 128;
  float inv = 1.0f / fmaxf(slot[0] * (1.0f / 127.0f), 1e-8f);
  for (size_t o = (size_t)blockIdx.x * blockDim.x + threadIdx.x; o < total;
       o += (size_t)gridDim.x * blockDim.x) {
    int d = (int)(o & 127);
    int s = (int)((o >> 7) & 2047);
    int hh = (int)((o >> 18) % nh);
    int b = (int)((o >> 18) / nh);
    float xv = x[((size_t)(b * 2048 + s) * nh + hh) * 128 + d];
    out[o] = (int8_t)clampi(__float2int_rn(xv * inv), -128, 127);
  }
}

__global__ void quantT_v_kernel(const float* __restrict__ x,
                                const float* __restrict__ slot,
                                int8_t* __restrict__ out) {
  size_t total = (size_t)2 * 8 * 128 * 2048;
  float inv = 1.0f / fmaxf(slot[0] * (1.0f / 127.0f), 1e-8f);
  for (size_t o = (size_t)blockIdx.x * blockDim.x + threadIdx.x; o < total;
       o += (size_t)gridDim.x * blockDim.x) {
    int s = (int)(o & 2047);
    int d = (int)((o >> 11) & 127);
    int hk = (int)((o >> 18) & 7);
    int b = (int)(o >> 21);
    float xv = x[(size_t)(b * 2048 + s) * 1024 + hk * 128 + d];
    out[o] = (int8_t)clampi(__float2int_rn(xv * inv), -128, 127);
  }
}

__global__ void attn_kernel(const int8_t* __restrict__ qq,
                            const int8_t* __restrict__ qk,
                            const int8_t* __restrict__ qvT,
                            const float* __restrict__ amax,
                            float* __restrict__ out) {
  const int SEQ = 2048, HD = 128;
  __shared__ char kS[64 * 128];
  __shared__ char vS[128 * 64];
  __shared__ char pS[8 * 16 * 64];
  const int wave = threadIdx.x >> 5, lane = threadIdx.x & 31;
  const int hi = lane >> 4, ln = lane & 15;
  const int b = blockIdx.z, h = blockIdx.y, qbase = blockIdx.x * 128;
  const int hkv = h >> 1;
  const int8_t* qptr = qq + (size_t)(b * 16 + h) * SEQ * HD;
  const int8_t* kptr = qk + (size_t)(b * 8 + hkv) * SEQ * HD;
  const int8_t* vptr = qvT + (size_t)(b * 8 + hkv) * HD * SEQ;
  float sq = fmaxf(amax[5] * (1.0f / 127.0f), 1e-8f);
  float sk = fmaxf(amax[6] * (1.0f / 127.0f), 1e-8f);
  float sv = fmaxf(amax[7] * (1.0f / 127.0f), 1e-8f);
  float sscale = sq * sk * 0.08838834764831845f;

  int qrow = qbase + wave * 16 + ln;
  v8i qa0 = load_a_tile(qptr + (size_t)qrow * HD, hi);
  v8i qa1 = load_a_tile(qptr + (size_t)qrow * HD + 64, hi);

  int nch_wave = (qbase + wave * 16 + 15) / 64 + 1;
  int nch_blk = (qbase + 127) / 64 + 1;

  float m[8], l[8];
#pragma unroll
  for (int r = 0; r < 8; ++r) { m[r] = -3.0e38f; l[r] = 0.0f; }

  for (int ch = 0; ch < nch_blk; ++ch) {
    int kb = ch * 64;
    {
      int row = threadIdx.x >> 2, q4 = threadIdx.x & 3;
      const char* src = (const char*)kptr + (size_t)(kb + row) * HD + q4 * 32;
      char* dst = kS + row * 128 + q4 * 32;
      *(uint4*)dst = *(const uint4*)src;
      *(uint4*)(dst + 16) = *(const uint4*)(src + 16);
      __builtin_prefetch(src + 64 * HD, 0, 3);
    }
    __syncthreads();
    if (ch < nch_wave) {
#pragma unroll
      for (int nt = 0; nt < 4; ++nt) {
        v8i acc = {};
        const char* krow = kS + (nt * 16 + ln) * 128;
        acc = wmma_iu8(qa0, load_b_tile(krow, hi), acc);
        acc = wmma_iu8(qa1, load_b_tile(krow + 64, hi), acc);
        int key = kb + nt * 16 + ln;
#pragma unroll
        for (int r = 0; r < 8; ++r) {
          int qi = qbase + wave * 16 + r + hi * 8;
          float s = (key <= qi) ? (float)acc[r] * sscale : -1.0e9f;
          float rmax = s;
#pragma unroll
          for (int off = 8; off; off >>= 1)
            rmax = fmaxf(rmax, __shfl_xor(rmax, off, 32));
          float mn = fmaxf(m[r], rmax);
          float e = expf(s - mn);
          float rsum = e;
#pragma unroll
          for (int off = 8; off; off >>= 1) rsum += __shfl_xor(rsum, off, 32);
          l[r] = l[r] * expf(m[r] - mn) + rsum;
          m[r] = mn;
        }
      }
    }
    __syncthreads();
  }

  float linv[8];
#pragma unroll
  for (int r = 0; r < 8; ++r) linv[r] = 1.0f / l[r];

  v8i o[8] = {};

  for (int ch = 0; ch < nch_blk; ++ch) {
    int kb = ch * 64;
    {
      int row = threadIdx.x >> 2, q4 = threadIdx.x & 3;
      const char* src = (const char*)kptr + (size_t)(kb + row) * HD + q4 * 32;
      char* dst = kS + row * 128 + q4 * 32;
      *(uint4*)dst = *(const uint4*)src;
      *(uint4*)(dst + 16) = *(const uint4*)(src + 16);
      __builtin_prefetch(src + 64 * HD, 0, 3);
    }
    {
      int d = threadIdx.x >> 1, half = threadIdx.x & 1;
      const char* src = (const char*)vptr + (size_t)d * SEQ + kb + half * 32;
      char* dst = vS + d * 64 + half * 32;
      *(uint4*)dst = *(const uint4*)src;
      *(uint4*)(dst + 16) = *(const uint4*)(src + 16);
      __builtin_prefetch(src + 64, 0, 3);
    }
    __syncthreads();
    if (ch < nch_wave) {
      char* pw = pS + wave * 16 * 64;
#pragma unroll
      for (int nt = 0; nt < 4; ++nt) {
        v8i acc = {};
        const char* krow = kS + (nt * 16 + ln) * 128;
        acc = wmma_iu8(qa0, load_b_tile(krow, hi), acc);
        acc = wmma_iu8(qa1, load_b_tile(krow + 64, hi), acc);
        int key = kb + nt * 16 + ln;
#pragma unroll
        for (int r = 0; r < 8; ++r) {
          int qi = qbase + wave * 16 + r + hi * 8;
          float s = (key <= qi) ? (float)acc[r] * sscale : -1.0e9f;
          float p = expf(s - m[r]) * linv[r];
          int qp = clampi(__float2int_rn(p * 127.0f), 0, 127);
          pw[(r + 8 * hi) * 64 + nt * 16 + ln] = (char)qp;
        }
      }
      v8i pa = load_a_tile(pw + ln * 64, hi);
#pragma unroll
      for (int t = 0; t < 8; ++t)
        o[t] = wmma_iu8(pa, load_b_tile(vS + (t * 16 + ln) * 64, hi), o[t]);
    }
    __syncthreads();
  }

  float osc = (1.0f / 127.0f) * sv;
  int row0 = qbase + wave * 16 + hi * 8;
#pragma unroll
  for (int t = 0; t < 8; ++t) {
    int col = h * 128 + t * 16 + ln;
#pragma unroll
    for (int r = 0; r < 8; ++r)
      out[(size_t)(b * SEQ + row0 + r) * 2048 + col] = (float)o[t][r] * osc;
  }
}

extern "C" void kernel_launch(void* const* d_in, const int* in_sizes, int n_in,
                              void* d_out, int out_size, void* d_ws,
                              size_t ws_size, hipStream_t stream) {
  (void)in_sizes; (void)n_in; (void)out_size; (void)ws_size;
  const float* hidden = (const float*)d_in[0];
  const float* cosb = (const float*)d_in[1];
  const float* sinb = (const float*)d_in[2];
  const float* wq = (const float*)d_in[4];
  const float* wk = (const float*)d_in[5];
  const float* wv = (const float*)d_in[6];
  const float* wo = (const float*)d_in[7];
  float* outp = (float*)d_out;

  const size_t N_HID = (size_t)2 * 2048 * 2048;
  const size_t N_WQ = (size_t)2048 * 2048;
  const size_t N_WKV = (size_t)2048 * 1024;
  const size_t N_KV = (size_t)4096 * 1024;
  const int M = 4096;

  char* ws = (char*)d_ws;
  size_t off = 0;
  auto alloc = [&](size_t bytes) {
    size_t o = off;
    off = (off + bytes + 255) & ~(size_t)255;
    return o;
  };
  float* amax = (float*)(ws + alloc(64));
  int8_t* qx = (int8_t*)(ws + alloc(N_HID));
  int8_t* qwq = (int8_t*)(ws + alloc(N_WQ));
  int8_t* qwk = (int8_t*)(ws + alloc(N_WKV));
  int8_t* qwv = (int8_t*)(ws + alloc(N_WKV));
  int8_t* qwo = (int8_t*)(ws + alloc(N_WQ));
  float* qf32 = (float*)(ws + alloc(N_HID * 4));
  float* kf32 = (float*)(ws + alloc(N_KV * 4));
  float* vf32 = (float*)(ws + alloc(N_KV * 4));
  int8_t* qqh = (int8_t*)(ws + alloc(N_HID));
  int8_t* qkh = (int8_t*)(ws + alloc(N_HID / 2));
  int8_t* qvT = (int8_t*)(ws + alloc(N_HID / 2));
  float* attn_out = qf32;
  int8_t* qattn = qx;

  zero_amax_kernel<<<1, 32, 0, stream>>>(amax);

  absmax_kernel<<<1024, 256, 0, stream>>>(hidden, N_HID, amax + 0);
  absmax_kernel<<<1024, 256, 0, stream>>>(wq, N_WQ, amax + 1);
  absmax_kernel<<<1024, 256, 0, stream>>>(wk, N_WKV, amax + 2);
  absmax_kernel<<<1024, 256, 0, stream>>>(wv, N_WKV, amax + 3);
  absmax_kernel<<<1024, 256, 0, stream>>>(wo, N_WQ, amax + 4);

  quant_kernel<<<4096, 256, 0, stream>>>(hidden, N_HID, amax + 0, qx);
  quant_kernel<<<4096, 256, 0, stream>>>(wq, N_WQ, amax + 1, qwq);
  quant_kernel<<<4096, 256, 0, stream>>>(wk, N_WKV, amax + 2, qwk);
  quant_kernel<<<4096, 256, 0, stream>>>(wv, N_WKV, amax + 3, qwv);
  quant_kernel<<<4096, 256, 0, stream>>>(wo, N_WQ, amax + 4, qwo);

  gemm_i8_kernel<<<dim3(2048 / 64, M / 128), 256, 0, stream>>>(
      qx, qwq, qf32, M, 2048, 2048, amax + 0, amax + 1);
  gemm_i8_kernel<<<dim3(1024 / 64, M / 128), 256, 0, stream>>>(
      qx, qwk, kf32, M, 1024, 2048, amax + 0, amax + 2);
  gemm_i8_kernel<<<dim3(1024 / 64, M / 128), 256, 0, stream>>>(
      qx, qwv, vf32, M, 1024, 2048, amax + 0, amax + 3);

  rope_absmax_kernel<<<2048, 256, 0, stream>>>(qf32, cosb, sinb, 16, amax + 5);
  rope_absmax_kernel<<<2048, 256, 0, stream>>>(kf32, cosb, sinb, 8, amax + 6);
  absmax_kernel<<<1024, 256, 0, stream>>>(vf32, N_KV, amax + 7);

  quantT_head_kernel<<<4096, 256, 0, stream>>>(qf32, amax + 5, 16, qqh);
  quantT_head_kernel<<<4096, 256, 0, stream>>>(kf32, amax + 6, 8, qkh);
  quantT_v_kernel<<<4096, 256, 0, stream>>>(vf32, amax + 7, qvT);

  attn_kernel<<<dim3(16, 16, 2), 256, 0, stream>>>(qqh, qkh, qvT, amax,
                                                   attn_out);

  absmax_kernel<<<1024, 256, 0, stream>>>(attn_out, N_HID, amax + 8);
  quant_kernel<<<4096, 256, 0, stream>>>(attn_out, N_HID, amax + 8, qattn);
  gemm_i8_kernel<<<dim3(2048 / 64, M / 128), 256, 0, stream>>>(
      qattn, qwo, outp, M, 2048, 2048, amax + 8, amax + 4);
}
